// kNN_Attention_efficient_13838384627983
// MI455X (gfx1250) — compile-verified, matrix operand order checked by reading, not run on hardware
//
#include <hip/hip_runtime.h>
#include <hip/hip_bf16.h>

// MI455X / gfx1250, wave32. Memory-bound problem (~26 GFLOP vs >=512MB HBM
// traffic -> ~22us floor @ 23.3TB/s). Strategy: algebraic fusion.
//   A = scale * Wv (X^T X) Wk^T      (Gram trick: no K/V materialization)
//   S = softmax(top50-mask(A))
//   out[i,j] = sum_j' U[i,j'] Wp[j,j'] + b[j] + x[i,j],
//     U[i,j'] = T[i>>10, (i&1023)*64 + j'],  T = S @ Q^T, Q recomputed on the fly.
// CDNA5 paths: v_wmma_f32_16x16x32_f16 for all GEMMs, async global->LDS
// double buffering (ASYNCcnt) in k_gram, TDM tensor_load_to_lds (TENSORcnt)
// staging in k_out, global_prefetch_b8 for the residual re-read.

typedef __attribute__((ext_vector_type(16))) _Float16 v16h;
typedef __attribute__((ext_vector_type(8)))  _Float16 v8h;
typedef __attribute__((ext_vector_type(4)))  _Float16 v4h;
typedef __attribute__((ext_vector_type(8)))  float    v8f;
typedef __attribute__((ext_vector_type(4)))  unsigned int u32x4;
typedef __attribute__((ext_vector_type(4)))  int      i32x4;
typedef __attribute__((ext_vector_type(8)))  int      i32x8;

#define B_      8
#define C_      64
#define N_      65536
#define SLICES  32
#define CHUNK   256
#define NCH     ((N_ / SLICES) / CHUNK)   // 8 chunks per slice

__device__ __forceinline__ v8f wmma16x16x32(v16h a, v16h b, v8f c) {
  return __builtin_amdgcn_wmma_f32_16x16x32_f16(false, a, false, b, (short)0, c,
                                                false, false);
}

// WMMA A/B fragment loader (ISA 7.12.2): element e -> k = (e&7) + (e>>3)*16 +
// (lane>>4)*8, row = lane&15. Caller passes p = row_base + kbase + (lane>>4)*8.
__device__ __forceinline__ v16h ld_frag(const _Float16* p) {
  v8h lo = *(const v8h*)(p);
  v8h hi = *(const v8h*)(p + 16);
  v16h r;
#pragma unroll
  for (int e = 0; e < 8; ++e) { r[e] = lo[e]; r[e + 8] = hi[e]; }
  return r;
}

// Async copy of one 64KB chunk (256 rows x 64 fp32) global -> LDS.
// 16 x b128 per lane; tracked by ASYNCcnt, completes in order.
__device__ __forceinline__ void async_copy_chunk(const float* gsrc,
                                                 const float* ldst, int t) {
#pragma unroll
  for (int q = 0; q < 16; ++q) {
    unsigned off = (unsigned)((q * 256 + t) * 16);
    unsigned lds_addr = (unsigned)(size_t)ldst + off;
    unsigned long long gaddr = (unsigned long long)(size_t)gsrc + (unsigned long long)off;
    asm volatile("global_load_async_to_lds_b128 %0, %1, off"
                 :: "v"(lds_addr), "v"(gaddr) : "memory");
  }
}

// ---------------------------------------------------------------------------
// Kernel 1: per-(batch, slice) partial Gram matrix  G_s = Xc^T Xc (64x64 f32)
// grid (SLICES, B), 256 threads (8 waves), 2 output 16x16 tiles per wave.
// Double-buffered async global->LDS pipeline hides HBM latency.
// Dynamic LDS: xraw 2 x 64KB fp32 | xt 64x264 f16  (~161 KB)
// ---------------------------------------------------------------------------
__global__ __launch_bounds__(256) void k_gram(const float* __restrict__ x,
                                              float* __restrict__ gpart) {
  extern __shared__ float smem1[];
  float*    xraw = smem1;                          // [2][CHUNK*64]
  _Float16* xt   = (_Float16*)(smem1 + 2 * CHUNK * 64);  // [64*264] padded X^T
  const int s = blockIdx.x;
  const int b = blockIdx.y;
  const int t = threadIdx.x;
  const int lane = t & 31, wave = t >> 5;
  const int tile0 = wave * 2, tile1 = wave * 2 + 1;
  const int tm0 = (tile0 >> 2) * 16, tn0 = (tile0 & 3) * 16;
  const int tm1 = (tile1 >> 2) * 16, tn1 = (tile1 & 3) * 16;
  const int m = lane & 15, half = lane >> 4;
  v8f acc0 = {}; v8f acc1 = {};
  const float* xb = x + ((size_t)b << 22) + (size_t)s * (N_ / SLICES) * C_;

  async_copy_chunk(xb, &xraw[0], t);               // prime the pipeline

  for (int ch = 0; ch < NCH; ++ch) {
    if (ch + 1 < NCH) {
      async_copy_chunk(xb + (size_t)(ch + 1) * CHUNK * C_,
                       &xraw[((ch + 1) & 1) * CHUNK * 64], t);
      asm volatile("s_wait_asynccnt 0x10" ::: "memory");  // oldest 16 done
    } else {
      asm volatile("s_wait_asynccnt 0x0" ::: "memory");
    }
    __syncthreads();   // all waves' async data for chunk ch visible

    // LDS fp32 -> transposed f16 xt[c][n]
    const float* src = &xraw[(ch & 1) * CHUNK * 64];
#pragma unroll
    for (int it = 0; it < 16; ++it) {
      int f = it * 1024 + t * 4;
      int n = f >> 6, c = f & 63;
      float4 v = *(const float4*)(src + f);
      xt[(c + 0) * 264 + n] = (_Float16)v.x;
      xt[(c + 1) * 264 + n] = (_Float16)v.y;
      xt[(c + 2) * 264 + n] = (_Float16)v.z;
      xt[(c + 3) * 264 + n] = (_Float16)v.w;
    }
    __syncthreads();

#pragma unroll
    for (int k = 0; k < CHUNK; k += 32) {
      v16h a0 = ld_frag(&xt[(tm0 + m) * 264 + k + half * 8]);
      v16h b0 = ld_frag(&xt[(tn0 + m) * 264 + k + half * 8]);
      acc0 = wmma16x16x32(a0, b0, acc0);
      v16h a1 = ld_frag(&xt[(tm1 + m) * 264 + k + half * 8]);
      v16h b1 = ld_frag(&xt[(tn1 + m) * 264 + k + half * 8]);
      acc1 = wmma16x16x32(a1, b1, acc1);
    }
  }
  float* gp = gpart + (size_t)(b * SLICES + s) * 4096;
  const int col = lane & 15;
#pragma unroll
  for (int r = 0; r < 8; ++r) {
    gp[(tm0 + r + half * 8) * 64 + tn0 + col] = acc0[r];
    gp[(tm1 + r + half * 8) * 64 + tn1 + col] = acc1[r];
  }
}

// ---------------------------------------------------------------------------
// Kernel 2: reduce Gram partials, A = scale * Wv G Wk^T (fp32),
// top-50 mask (drop 14 smallest) + softmax, store S as f16.  grid (B), 256 thr.
// ---------------------------------------------------------------------------
__global__ __launch_bounds__(256) void k_attn(const float* __restrict__ gpart,
                                              const float* __restrict__ wqkv,
                                              _Float16* __restrict__ Sout) {
  __shared__ float G[4096];
  __shared__ float M1[4096];
  __shared__ float A[4096];
  const int b = blockIdx.x;
  const int t = threadIdx.x;

  for (int i = t; i < 4096; i += 256) {
    float acc = 0.f;
    for (int s = 0; s < SLICES; ++s)
      acc += gpart[(size_t)(b * SLICES + s) * 4096 + i];
    G[i] = acc;
  }
  __syncthreads();
  for (int idx = t; idx < 4096; idx += 256) {     // M1 = G @ Wk^T
    int i = idx >> 6, d = idx & 63;
    float acc = 0.f;
    for (int j = 0; j < 64; ++j) acc += G[i * 64 + j] * wqkv[(64 + d) * 64 + j];
    M1[idx] = acc;
  }
  __syncthreads();
  for (int idx = t; idx < 4096; idx += 256) {     // A = scale * Wv @ M1
    int c = idx >> 6, d = idx & 63;
    float acc = 0.f;
    for (int i = 0; i < 64; ++i) acc += wqkv[(128 + c) * 64 + i] * M1[i * 64 + d];
    A[idx] = acc * 0.125f;
  }
  __syncthreads();
  if (t < 64) {   // per-row: drop 14 smallest == keep top 50, then softmax
    const float* row = &A[t * 64];
    unsigned long long drop = 0ull;
    for (int it = 0; it < 14; ++it) {
      float mn = 3.4e38f; int mi = 0;
      for (int j = 0; j < 64; ++j)
        if (!((drop >> j) & 1ull) && row[j] < mn) { mn = row[j]; mi = j; }
      drop |= (1ull << mi);
    }
    float mx = -3.4e38f;
    for (int j = 0; j < 64; ++j)
      if (!((drop >> j) & 1ull)) mx = fmaxf(mx, row[j]);
    float sum = 0.f;
    for (int j = 0; j < 64; ++j)
      if (!((drop >> j) & 1ull)) sum += __expf(row[j] - mx);
    float inv = 1.f / sum;
    _Float16* srow = Sout + ((size_t)b << 12) + t * 64;
    for (int j = 0; j < 64; ++j) {
      float v = ((drop >> j) & 1ull) ? 0.f : __expf(row[j] - mx) * inv;
      srow[j] = (_Float16)v;
    }
  }
}

// ---------------------------------------------------------------------------
// Kernel 3: fused  Q = X Wq^T  ->  T = S Q^T  ->  reshape -> Y = U Wp^T + b + x
// grid (256 r-blocks, B).  Block handles r in [rb*4, rb*4+4) for ALL 64 c,
// i.e. Q rows n in [rb*256, rb*256+256).  256 threads = 8 waves.
// X slab staged by the Tensor Data Mover (one 1-D descriptor, TENSORcnt).
// ---------------------------------------------------------------------------
#if __has_builtin(__builtin_amdgcn_tensor_load_to_lds)
#warning "CDNA5 probe: tensor_load_to_lds builtin PRESENT - TDM path enabled"
#if defined(__clang_major__) && (__clang_major__ >= 23)
#warning "CDNA5 probe: using 6-arg tensor_load_to_lds (clang>=23)"
#else
#warning "CDNA5 probe: using 5-arg tensor_load_to_lds (clang<23)"
#endif
#else
#warning "CDNA5 probe: tensor_load_to_lds builtin ABSENT - fallback path"
#endif

__global__ __launch_bounds__(256) void k_out(const float* __restrict__ x,
                                             const float* __restrict__ wqkv,
                                             const float* __restrict__ wproj,
                                             const float* __restrict__ bproj,
                                             const _Float16* __restrict__ S,
                                             float* __restrict__ out) {
  extern __shared__ _Float16 smem[];
  _Float16* Xs  = smem;                 // 256*72
  _Float16* Qs  = smem + 18432;         // 256*72
  _Float16* Tl  = smem + 36864;         // 64*264
  _Float16* Ss  = smem + 53760;         // 64*72
  _Float16* Wqs = smem + 58368;         // 64*72
  _Float16* Wps = smem + 62976;         // 64*72
  float*    Xraw = (float*)(smem + 67584);  // 256*64 fp32 TDM landing pad

  const int rb = blockIdx.x;            // 0..255
  const int b  = blockIdx.y;
  const int t  = threadIdx.x, lane = t & 31, wave = t >> 5;
  const int n0 = rb * 256;
  const size_t xbase = (size_t)b << 22;
  const int m = lane & 15, half = lane >> 4;

  // ---- Phase 0: TDM-issue the X slab, stage weights while the DMA runs ----
#if __has_builtin(__builtin_amdgcn_tensor_load_to_lds)
  if (t < 32) {  // wave 0 issues one descriptor (TENSOR ops ignore EXEC)
    unsigned lds_addr = (unsigned)(size_t)Xraw;
    unsigned long long ga =
        (unsigned long long)(size_t)(x + xbase + (size_t)n0 * 64);
    // D# group0: count=1 | lds_addr[63:32] | global_addr[120:64] | type=2
    u32x4 g0 = {1u, lds_addr, (unsigned)ga,
                (unsigned)((ga >> 32) & 0x01FFFFFFu) | 0x80000000u};
    // D# group1: data_size=4B, tensor_dim0=16384, tile_dim0=16384, stride=16384
    i32x8 g1 = {0x00020000, 0x40000000, 0, 0x40000000, 0, 16384, 0, 0};
    i32x4 gz = {0, 0, 0, 0};
#if defined(__clang_major__) && (__clang_major__ >= 23)
    i32x8 gz8 = {0, 0, 0, 0, 0, 0, 0, 0};
    __builtin_amdgcn_tensor_load_to_lds(g0, g1, gz, gz, gz8, 0);
#else
    __builtin_amdgcn_tensor_load_to_lds(g0, g1, gz, gz, 0);
#endif
  }
#endif
  // Weights + S staging overlaps with the tensor DMA.
  for (int idx = t; idx < 4096; idx += 256) {
    int r = idx >> 6, c = idx & 63;
    Wqs[r * 72 + c] = (_Float16)wqkv[idx];        // Wq = rows 0..63
    Wps[r * 72 + c] = (_Float16)wproj[idx];
    Ss[r * 72 + c]  = S[((size_t)b << 12) + idx];
  }
  // Prefetch the residual rows we re-read in phase 3 (scattered by c).
  {
    const float* pr =
        x + xbase + ((size_t)(t >> 2) * 1024 + rb * 4 + (t & 3)) * 64;
    __builtin_prefetch(pr, 0, 0);
    __builtin_prefetch(pr + 32, 0, 0);
  }
#if __has_builtin(__builtin_amdgcn_tensor_load_to_lds)
  if (t < 32) __builtin_amdgcn_s_wait_tensorcnt(0);
  __syncthreads();
  // LDS fp32 -> f16 row-major Xs[n][c]
#pragma unroll
  for (int it = 0; it < 16; ++it) {
    int f = it * 1024 + t * 4;
    int n = f >> 6, c = f & 63;
    float4 v = *(const float4*)(Xraw + f);
    v4h h; h[0] = (_Float16)v.x; h[1] = (_Float16)v.y;
    h[2] = (_Float16)v.z; h[3] = (_Float16)v.w;
    *(v4h*)(&Xs[n * 72 + c]) = h;
  }
#else
  {  // fallback: direct global fp32 -> f16
    const float* xr = x + xbase + (size_t)(n0 + t) * 64;
    _Float16* dst = &Xs[t * 72];
#pragma unroll
    for (int c = 0; c < 64; c += 4) {
      float4 v = *(const float4*)(xr + c);
      v4h h; h[0] = (_Float16)v.x; h[1] = (_Float16)v.y;
      h[2] = (_Float16)v.z; h[3] = (_Float16)v.w;
      *(v4h*)(dst + c) = h;
    }
  }
#endif
  __syncthreads();

  // ---- Phase 1: Q[n,d] = sum_c Xs[n,c] Wq[d,c].  64 tiles -> 8 per wave ----
#pragma unroll
  for (int i = 0; i < 8; ++i) {
    int tq = wave * 8 + i, nt = tq >> 2, dt = tq & 3;
    v8f acc = {};
#pragma unroll
    for (int k = 0; k < 64; k += 32) {
      v16h a  = ld_frag(&Xs[(nt * 16 + m) * 72 + k + half * 8]);
      v16h bf = ld_frag(&Wqs[(dt * 16 + m) * 72 + k + half * 8]);
      acc = wmma16x16x32(a, bf, acc);
    }
#pragma unroll
    for (int r = 0; r < 8; ++r)
      Qs[(nt * 16 + r + half * 8) * 72 + dt * 16 + m] = (_Float16)acc[r];
  }
  __syncthreads();

  // ---- Phase 2: T[c,nl] = sum_d Ss[c,d] Qs[nl,d].  4x16 tiles ----
#pragma unroll
  for (int i = 0; i < 8; ++i) {
    int tt = wave * 8 + i, cm = tt >> 4, nt = tt & 15;
    v8f acc = {};
#pragma unroll
    for (int k = 0; k < 64; k += 32) {
      v16h a  = ld_frag(&Ss[(cm * 16 + m) * 72 + k + half * 8]);
      v16h bf = ld_frag(&Qs[(nt * 16 + m) * 72 + k + half * 8]);
      acc = wmma16x16x32(a, bf, acc);
    }
#pragma unroll
    for (int r = 0; r < 8; ++r)
      Tl[(cm * 16 + r + half * 8) * 264 + nt * 16 + m] = (_Float16)acc[r];
  }
  __syncthreads();

  // ---- Phase 3: U[u=c*4+rl, j'] = Tl[c][rl*64+j'];  Y = U Wp^T + b + x ----
#pragma unroll
  for (int i = 0; i < 8; ++i) {
    int ty = wave * 8 + i, mt = ty >> 2, ntj = ty & 3;
    v8f acc = {};
#pragma unroll
    for (int k = 0; k < 64; k += 32) {
      int u = mt * 16 + m;
      v16h a  = ld_frag(&Tl[(u >> 2) * 264 + (u & 3) * 64 + k + half * 8]);
      v16h bf = ld_frag(&Wps[(ntj * 16 + m) * 72 + k + half * 8]);
      acc = wmma16x16x32(a, bf, acc);
    }
    int j = ntj * 16 + m;
    float bj = bproj[j];
#pragma unroll
    for (int r = 0; r < 8; ++r) {
      int u = mt * 16 + r + half * 8;
      int c = u >> 2, rl = u & 3;
      size_t o = xbase + ((size_t)c * 1024 + rb * 4 + rl) * 64 + j;
      out[o] = acc[r] + bj + x[o];
    }
  }
}

// ---------------------------------------------------------------------------
extern "C" void kernel_launch(void* const* d_in, const int* in_sizes, int n_in,
                              void* d_out, int out_size, void* d_ws, size_t ws_size,
                              hipStream_t stream) {
  const float* x     = (const float*)d_in[0];
  const float* wqkv  = (const float*)d_in[1];
  const float* wproj = (const float*)d_in[2];
  const float* bproj = (const float*)d_in[3];
  float* out = (float*)d_out;

  // workspace: [0, 4MB) Gram partials (B*SLICES*64*64 f32); [4MB, +64KB) S f16
  float*     gpart = (float*)d_ws;
  _Float16*  Smat  = (_Float16*)((char*)d_ws + (size_t)B_ * SLICES * 4096 * 4);

  const size_t lds1 = (2 * CHUNK * 64) * sizeof(float) + (64 * 264) * sizeof(_Float16);
  const size_t lds3 = 100352 * sizeof(_Float16);  // 67584 f16 + 64KB fp32 pad

  k_gram<<<dim3(SLICES, B_), 256, lds1, stream>>>(x, gpart);
  k_attn<<<dim3(B_), 256, 0, stream>>>(gpart, wqkv, Smat);
  k_out<<<dim3(256, B_), 256, lds3, stream>>>(x, wqkv, wproj, bproj, Smat, out);
}
